// Encoder_53171695125330
// MI455X (gfx1250) — compile-verified, matrix operand order checked by reading, not run on hardware
//
#include <hip/hip_runtime.h>

// ---------------------------------------------------------------------------
// Encoder for gfx1250 (MI455X).
//  out layout: [40*120*120] final_spatial, then [4] scalar MLP  (576004 f32)
//  ws layout : Wn   (128x800 f32, normalized triangle weights, rows>=120 zero)
//              Wrev (128x800 f32, Wn with reversed source axis,  rows>=120 zero)
//              tmpT (8 x 128 x 800 f32, pass-1 GEMM output, [c][oy][u])
//  total ws: ~4.10 MB
// ---------------------------------------------------------------------------

typedef __attribute__((ext_vector_type(16))) _Float16 v16h;
typedef __attribute__((ext_vector_type(8)))  float    v8f;
typedef __attribute__((ext_vector_type(4)))  float    v4f;
typedef __attribute__((ext_vector_type(4)))  int      v4i;

#define SCALE 0.15f         // 120/800
#define OUT_HW 14400        // 120*120
#define WROWS 128           // padded weight rows

#if defined(__has_builtin)
#if __has_builtin(__builtin_amdgcn_global_load_async_to_lds_b128) && \
    __has_builtin(__builtin_amdgcn_s_wait_asynccnt)
#define HAVE_ASYNC_LDS 1
#endif
#endif

// nan_to_num(posinf=0) -> clamp(>255) -> /255 ; keep f16-finite.  Branchless.
__device__ __forceinline__ float sanitize(float x) {
    x = (x <= 3.0e38f) ? x : 0.0f;          // NaN and +inf -> 0
    x = fmaxf(x, -3.0e38f);                 // -inf -> large negative
    x = fminf(x, 255.0f);
    x *= (1.0f / 255.0f);
    return fmaxf(x, -60000.0f);             // stay finite in f16
}

// --- build normalized antialiased triangle-filter weights (jax.image.resize)
//     rows 120..127 are zero padding so GEMM fragments never predicate.
__global__ void build_weights(float* __restrict__ Wn, float* __restrict__ Wrev) {
    int o = threadIdx.x;
    if (o >= WROWS) return;
    if (o >= 120) {
        for (int x = 0; x < 800; ++x) { Wn[o * 800 + x] = 0.0f; Wrev[o * 800 + x] = 0.0f; }
        return;
    }
    float c = (o + 0.5f) / SCALE - 0.5f;    // output center in source coords
    float sum = 0.0f;
    for (int x = 0; x < 800; ++x) {
        float t = SCALE * ((float)x - c);
        float w = 1.0f - fabsf(t);
        w = w > 0.0f ? w : 0.0f;
        Wn[o * 800 + x] = w;
        sum += w;
    }
    float inv = 1.0f / sum;
    for (int x = 0; x < 800; ++x) {
        float w = Wn[o * 800 + x] * inv;
        Wn[o * 800 + x] = w;
        Wrev[o * 800 + (799 - x)] = w;      // folds rot90 into the weights
    }
}

// --- zero the scatter channels (8..39) -- atomics accumulate into them
__global__ void zero_kernel(float* __restrict__ out) {
    int i = blockIdx.x * blockDim.x + threadIdx.x;
    if (i < 8 * OUT_HW) ((v4f*)(out + 8 * OUT_HW))[i] = (v4f){0.f, 0.f, 0.f, 0.f};
}

// Pack one f16 A/B fragment (CDNA5 16-bit 16x32 layout) from a row pointer.
// Lane-half hh holds K = hh*8+0..7 (VGPRs 0-3) and K = 16+hh*8+0..7 (VGPRs 4-7):
// two contiguous 8-float groups -> four b128 loads (global or ds).
__device__ __forceinline__ v16h pack_frag(const float* rowp, int k0, int hh) {
    const v4f* p0 = (const v4f*)(rowp + k0 + hh * 8);
    const v4f* p1 = (const v4f*)(rowp + k0 + 16 + hh * 8);
    v4f x0 = p0[0], x1 = p0[1], x2 = p1[0], x3 = p1[1];
    v16h f;
#pragma unroll
    for (int i = 0; i < 4; ++i) {
        f[i]      = (_Float16)x0[i];
        f[4 + i]  = (_Float16)x1[i];
        f[8 + i]  = (_Float16)x2[i];
        f[12 + i] = (_Float16)x3[i];
    }
    return f;
}

// Same, but sanitize (nan_to_num / clamp / scale) -- for the spatial input.
// Spatial rows are read once per oy-tile (8 times): nontemporal loads.
__device__ __forceinline__ v16h pack_frag_sanitize(const float* __restrict__ rowp, int k0, int hh) {
    const v4f* p0 = (const v4f*)(rowp + k0 + hh * 8);
    const v4f* p1 = (const v4f*)(rowp + k0 + 16 + hh * 8);
    v4f x0 = __builtin_nontemporal_load(p0);
    v4f x1 = __builtin_nontemporal_load(p0 + 1);
    v4f x2 = __builtin_nontemporal_load(p1);
    v4f x3 = __builtin_nontemporal_load(p1 + 1);
    v16h f;
#pragma unroll
    for (int i = 0; i < 4; ++i) {
        f[i]      = (_Float16)sanitize(x0[i]);
        f[4 + i]  = (_Float16)sanitize(x1[i]);
        f[8 + i]  = (_Float16)sanitize(x2[i]);
        f[12 + i] = (_Float16)sanitize(x3[i]);
    }
    return f;
}

// --- Pass 1: tmpT[c][oy][u] = sum_v P[c][u][v] * Wrev[oy][v]
//     GEMM  M=u (800, 25 groups of 2 tiles)  N=oy (128 pad, 8 tiles)  K=v (800)
//     Both waves of a block share one oy-tile: its 16 weight rows (51.2 KB)
//     are staged into LDS with async global->LDS copies (ASYNCcnt).
__global__ void __launch_bounds__(64) resize_pass1(const float* __restrict__ spatial,
                                                   const float* __restrict__ Wrev,
                                                   float* __restrict__ tmpT) {
    __shared__ float lw[16 * 800];          // 51.2 KB: weight rows n0..n0+15

    const int blk   = blockIdx.x;           // 8c * 8nt * 25mg = 1600 blocks
    const int c     = blk / 200;
    const int r     = blk % 200;
    const int ntile = r / 25;
    const int mg    = r % 25;
    const int tid   = threadIdx.x;          // 0..63 (2 waves)
    const int m0    = (mg * 2 + (tid >> 5)) * 16;   // u base (per wave)
    const int n0    = ntile * 16;                   // oy base (per block)
    const int lane  = tid & 31;
    const int hh    = lane >> 4;
    const int lrow  = lane & 15;

    // ---- stage 16x800 weight rows into LDS ----
    const float* wsrc = Wrev + (size_t)n0 * 800;
#ifdef HAVE_ASYNC_LDS
    for (int i = tid; i < 3200; i += 64)    // 3200 16-byte chunks
        __builtin_amdgcn_global_load_async_to_lds_b128(
            (v4i*)(wsrc + i * 4), (v4i*)(lw + i * 4), 0, 0);
    __builtin_amdgcn_s_wait_asynccnt(0);
#else
    for (int i = tid; i < 3200; i += 64)
        ((v4f*)lw)[i] = ((const v4f*)wsrc)[i];
#endif
    __syncthreads();

    const float* prow  = spatial + (size_t)c * 640000 + (size_t)(m0 + lrow) * 800;
    const float* lwrow = lw + (size_t)lrow * 800;   // B row oy = n0 + lrow (from LDS)

    v8f acc = {};
    for (int kb = 0; kb < 25; ++kb) {
        const int k0 = kb * 32;
        v16h a = pack_frag_sanitize(prow, k0, hh);
        v16h b = pack_frag(lwrow, k0, hh);          // ds_load_b128 x4
        acc = __builtin_amdgcn_wmma_f32_16x16x32_f16(false, a, false, b,
                                                     (short)0, acc, false, false);
    }
    // D layout: lane holds rows u = m0 + hh*8 + r, col oy = n0 + lrow.
    // Transposed store -> 8 u-contiguous floats per lane -> two b128 stores.
    float* trow = tmpT + ((size_t)c * WROWS + (n0 + lrow)) * 800 + m0 + hh * 8;
    ((v4f*)trow)[0] = (v4f){acc[0], acc[1], acc[2], acc[3]};
    ((v4f*)trow)[1] = (v4f){acc[4], acc[5], acc[6], acc[7]};
}

// --- Pass 2: out[c][oy][ox] = sum_u Wn[ox][u] * tmpT[c][oy][u]
//     GEMM  M=ox (128 pad, 8 tiles)  N=oy (128 pad, 8 tiles)  K=u (800)
__global__ void resize_pass2(const float* __restrict__ tmpT,
                             const float* __restrict__ Wn,
                             float* __restrict__ out) {
    const int wave = (blockIdx.x * blockDim.x + threadIdx.x) >> 5; // 512 waves exact
    const int c    = wave / 64;
    const int rem  = wave % 64;
    const int m0   = (rem / 8) * 16;        // ox base
    const int n0   = (rem % 8) * 16;        // oy base
    const int lane = threadIdx.x & 31;
    const int hh   = lane >> 4;
    const int lrow = lane & 15;

    const float* wrow = Wn + (size_t)(m0 + lrow) * 800;                       // padded
    const float* trow = tmpT + ((size_t)c * WROWS + (n0 + lrow)) * 800;       // padded

    v8f acc = {};
    for (int kb = 0; kb < 25; ++kb) {
        const int k0 = kb * 32;
        v16h a = pack_frag(wrow, k0, hh);
        v16h b = pack_frag(trow, k0, hh);
        acc = __builtin_amdgcn_wmma_f32_16x16x32_f16(false, a, false, b,
                                                     (short)0, acc, false, false);
    }
    // D[ox][oy]: lane holds oy = n0 + lrow, ox = m0 + hh*8 + 0..7 (contiguous).
    const int oy = n0 + lrow;
    if (oy < 120) {
        float* orow = out + c * OUT_HW + oy * 120;
        const int base = m0 + hh * 8;
        if (base + 3 < 120) ((v4f*)(orow + base))[0] = (v4f){acc[0], acc[1], acc[2], acc[3]};
        if (base + 7 < 120) ((v4f*)(orow + base))[1] = (v4f){acc[4], acc[5], acc[6], acc[7]};
    }
}

// --- Scatter: one wave per entity, lane = embedding channel.
//     Direct low-res scatter with resize-adjoint weights (<= 3x3 outputs).
__global__ void scatter_kernel(const float* __restrict__ emb,
                               const int*   __restrict__ loc,
                               const float* __restrict__ Wn,
                               float* __restrict__ out) {
    const int wave = (blockIdx.x * blockDim.x + threadIdx.x) >> 5;
    if (wave >= 100000) return;
    const int lane = threadIdx.x & 31;
    int x = loc[2 * wave];     x = min(max(x, 0), 799);
    int y = loc[2 * wave + 1]; y = min(max(y, 0), 799);
    const float e = __builtin_nontemporal_load(emb + (size_t)wave * 32 + lane);
    float* chan = out + (size_t)(8 + lane) * OUT_HW;

    const float cy = SCALE * ((float)y + 0.5f) - 0.5f;
    const float cx = SCALE * ((float)x + 0.5f) - 0.5f;
    const int oy0 = (int)floorf(cy) - 1;
    const int ox0 = (int)floorf(cx) - 1;
#pragma unroll
    for (int a = 0; a < 3; ++a) {
        const int oy = oy0 + a;
        if (oy < 0 || oy >= 120) continue;
        const float wy = Wn[oy * 800 + y];
        if (wy == 0.0f) continue;
#pragma unroll
        for (int bq = 0; bq < 3; ++bq) {
            const int ox = ox0 + bq;
            if (ox < 0 || ox >= 120) continue;
            const float wx = Wn[ox * 800 + x];
            if (wx == 0.0f) continue;
            atomicAdd(chan + oy * 120 + ox, e * wy * wx);
        }
    }
}

// --- tiny 8->32->4 ReLU MLP
__global__ void mlp_kernel(const float* __restrict__ sc,
                           const float* __restrict__ w1, const float* __restrict__ b1,
                           const float* __restrict__ w2, const float* __restrict__ b2,
                           float* __restrict__ out) {
    __shared__ float h[32];
    const int j = threadIdx.x;
    float acc = b1[j];
    for (int i = 0; i < 8; ++i) acc += sc[i] * w1[i * 32 + j];
    h[j] = acc > 0.0f ? acc : 0.0f;
    __syncthreads();
    if (j < 4) {
        float o = b2[j];
        for (int q = 0; q < 32; ++q) o += h[q] * w2[q * 4 + j];
        out[40 * OUT_HW + j] = o > 0.0f ? o : 0.0f;
    }
}

extern "C" void kernel_launch(void* const* d_in, const int* in_sizes, int n_in,
                              void* d_out, int out_size, void* d_ws, size_t ws_size,
                              hipStream_t stream) {
    (void)in_sizes; (void)n_in; (void)out_size; (void)ws_size;
    const float* spatial = (const float*)d_in[0];
    const float* emb     = (const float*)d_in[1];
    const int*   loc     = (const int*)d_in[2];
    const float* sc      = (const float*)d_in[3];
    const float* w1      = (const float*)d_in[4];
    const float* b1      = (const float*)d_in[5];
    const float* w2      = (const float*)d_in[6];
    const float* b2      = (const float*)d_in[7];
    float* out  = (float*)d_out;
    float* Wn   = (float*)d_ws;          // 128*800 f32
    float* Wrev = Wn + WROWS * 800;      // 128*800 f32
    float* tmpT = Wrev + WROWS * 800;    // 8*128*800 f32

    build_weights<<<1, 128, 0, stream>>>(Wn, Wrev);
    zero_kernel<<<450, 256, 0, stream>>>(out);                     // 115200 v4f
    resize_pass1<<<1600, 64, 0, stream>>>(spatial, Wrev, tmpT);    // 3200 waves
    resize_pass2<<<128, 128, 0, stream>>>(tmpT, Wn, out);          //  512 waves
    scatter_kernel<<<12500, 256, 0, stream>>>(emb, loc, Wn, out);  // 100000 waves
    mlp_kernel<<<1, 32, 0, stream>>>(sc, w1, b1, w2, b2, out);
}
